// MultiheadAttention_1168231105031
// MI455X (gfx1250) — matrix operand order repaired and checked by reading, not compiled, not run on hardware
//
#include <hip/hip_runtime.h>
#include <hip/hip_bf16.h>
#include <stdint.h>

#define BB  4
#define SS  2048
#define DD  1024
#define HH  16
#define HSZ 64
#define LN_EPS 1e-5f
#define ATT_SCALE 0.03125f   // D^-0.5 = 1024^-0.5

typedef __bf16        v16bf __attribute__((ext_vector_type(16)));
typedef float         v8f   __attribute__((ext_vector_type(8)));
typedef unsigned int  v4u   __attribute__((ext_vector_type(4)));
typedef int           v8i   __attribute__((ext_vector_type(8)));
typedef int           v4i   __attribute__((ext_vector_type(4)));

union Frag {
    v16bf v;
    uint4 q[2];
};

__device__ __forceinline__ unsigned short f2bf(float f) {
    union { float f; unsigned int u; } c; c.f = f;
    unsigned int u = c.u;
    u += 0x7FFFu + ((u >> 16) & 1u);   // round-to-nearest-even
    return (unsigned short)(u >> 16);
}

__device__ __forceinline__ v8f wmma_bf16(const Frag& a, const Frag& b, v8f c) {
    return __builtin_amdgcn_wmma_f32_16x16x32_bf16(false, a.v, false, b.v,
                                                   (short)0, c, false, false);
}

// ---------------------------------------------------------------------------
// Tensor Data Mover: 2D bf16 tile load global->LDS (ISA 08_async_tensor §8).
// Builds D# group0/group1 (remaining groups zero => 2D tensor). pad16 inserts
// 8 DWORDs (16 halves) after every 32 DWORDs (64 halves) stored, i.e. it
// realizes the 80-half LDS row stride used by the WMMA fragment loaders.
// This toolchain's builtin is the 6-arg form:
//   (uint32x4 g0, int32x8 g1, int32x4, int32x4, int32x8, i32 cpol)
// ---------------------------------------------------------------------------
__device__ __forceinline__ void tdm_load_tile_bf16(unsigned lds_off,
                                                   const unsigned short* gptr,
                                                   unsigned tile_w, unsigned tile_h,
                                                   unsigned row_stride_elems,
                                                   bool pad16) {
    unsigned long long ga = (unsigned long long)(uintptr_t)gptr;
    v4u g0 = { 1u,                                        // count=1, user D#
               lds_off,                                   // lds_addr
               (unsigned)(ga & 0xFFFFFFFFu),              // global_addr[31:0]
               ((unsigned)((ga >> 32) & 0x01FFFFFFu)) | (2u << 30) };  // [56:32] | type=2
    unsigned flags = (1u << 16);                          // data_size = 2 bytes
    if (pad16) flags |= (1u << 20)                        // pad_enable
                     |  (4u << 22)                        // pad_interval: 32 DWORDs
                     |  (7u << 25);                       // pad_amount:   8 DWORDs
    v8i g1;
    g1[0] = (int)flags;                                   // wg_mask=0 | size | pad
    g1[1] = (int)((tile_w & 0xFFFFu) << 16);              // tensor_dim0[15:0]
    g1[2] = (int)((tile_h & 0xFFFFu) << 16);              // dim0 hi=0 | tensor_dim1 lo
    g1[3] = (int)((tile_w & 0xFFFFu) << 16);              // dim1 hi=0 | tile_dim0
    g1[4] = (int)(tile_h & 0xFFFFu);                      // tile_dim1 | tile_dim2=0
    g1[5] = (int)row_stride_elems;                        // tensor_dim0_stride[31:0]
    g1[6] = 0;                                            // stride0 hi | stride1 lo
    g1[7] = 0;
    v4i z4 = { 0, 0, 0, 0 };
    v8i z8 = { 0, 0, 0, 0, 0, 0, 0, 0 };
    __builtin_amdgcn_tensor_load_to_lds(g0, g1, z4, z4, z8, 0);
}

// ---------------------------------------------------------------------------
// Kernel 1: fused LayerNorm (fp32 in) -> bf16 out, one block per row
// ---------------------------------------------------------------------------
__global__ void __launch_bounds__(256)
ln_kernel(const float* __restrict__ v, const float* __restrict__ k,
          const float* __restrict__ q,
          const float* __restrict__ g, const float* __restrict__ beta,
          unsigned short* __restrict__ vn, unsigned short* __restrict__ kn,
          unsigned short* __restrict__ qn) {
    const int row   = blockIdx.x;
    const int which = blockIdx.y;
    const float* src = (which == 0) ? v : (which == 1) ? k : q;
    unsigned short* dst = (which == 0) ? vn : (which == 1) ? kn : qn;

    const int tid = threadIdx.x;              // 256 threads, 4 floats each
    const float4 x = ((const float4*)(src + (size_t)row * DD))[tid];

    float s1 = x.x + x.y + x.z + x.w;
    float s2 = x.x * x.x + x.y * x.y + x.z * x.z + x.w * x.w;
    for (int i = 1; i < 32; i <<= 1) {
        s1 += __shfl_xor(s1, i, 32);
        s2 += __shfl_xor(s2, i, 32);
    }
    __shared__ float ws1[8], ws2[8];
    const int wid = tid >> 5, lane = tid & 31;
    if (lane == 0) { ws1[wid] = s1; ws2[wid] = s2; }
    __syncthreads();
    float t1 = 0.f, t2 = 0.f;
    for (int i = 0; i < 8; ++i) { t1 += ws1[i]; t2 += ws2[i]; }

    const float mu  = t1 * (1.0f / DD);
    const float var = t2 * (1.0f / DD) - mu * mu;
    const float rs  = rsqrtf(var + LN_EPS);

    const int col = tid * 4;
    const float y0 = (x.x - mu) * rs * g[col + 0] + beta[col + 0];
    const float y1 = (x.y - mu) * rs * g[col + 1] + beta[col + 1];
    const float y2 = (x.z - mu) * rs * g[col + 2] + beta[col + 2];
    const float y3 = (x.w - mu) * rs * g[col + 3] + beta[col + 3];

    uint2 pk;
    pk.x = (unsigned int)f2bf(y0) | ((unsigned int)f2bf(y1) << 16);
    pk.y = (unsigned int)f2bf(y2) | ((unsigned int)f2bf(y3) << 16);
    *(uint2*)(dst + (size_t)row * DD + col) = pk;
}

// ---------------------------------------------------------------------------
// Kernel 2: per-head QKV projection  [8192 x 1024] x [1024 x 64] (bf16 WMMA)
// A-tile staged by the Tensor Data Mover; W^T tile staged with VALU convert.
// ---------------------------------------------------------------------------
__global__ void __launch_bounds__(256)
proj_kernel(const unsigned short* __restrict__ qn, const unsigned short* __restrict__ kn,
            const unsigned short* __restrict__ vn,
            const float* __restrict__ Wq, const float* __restrict__ Wk,
            const float* __restrict__ Wv,
            unsigned short* __restrict__ qh, unsigned short* __restrict__ kh,
            unsigned short* __restrict__ vh) {
    const int which = blockIdx.z;
    const unsigned short* A = (which == 0) ? qn : (which == 1) ? kn : vn;
    const float*          W = (which == 0) ? Wq : (which == 1) ? Wk : Wv;
    unsigned short*     Dst = (which == 0) ? qh : (which == 1) ? kh : vh;

    const int h       = blockIdx.y;
    const int rowBase = blockIdx.x * 128;
    const int tid = threadIdx.x;
    const int wid = tid >> 5, lane = tid & 31;
    const int lo = lane & 15, hi = lane >> 4;

    __shared__ unsigned short lsA[128 * 80];   // activations tile [row][k] (TDM-padded)
    __shared__ unsigned short lsB[64 * 80];    // W^T tile [e][k]

    v8f acc[4];
#pragma unroll
    for (int n = 0; n < 4; ++n)
#pragma unroll
        for (int i = 0; i < 8; ++i) acc[n][i] = 0.f;

    for (int kb = 0; kb < DD; kb += 64) {
        // async DMA of the 128x64 activation tile (wave 0 drives the TDM)
        if (wid == 0)
            tdm_load_tile_bf16((unsigned)(uintptr_t)lsA,
                               A + (size_t)rowBase * DD + kb,
                               64, 128, DD, true);
        // stage W transposed + fp32->bf16: W[h][kb+d][e] -> lsB[e][d]
        for (int idx = tid; idx < 4096; idx += 256) {
            const int e = idx & 63, d = idx >> 6;
            lsB[e * 80 + d] = f2bf(W[((size_t)h * DD + (kb + d)) * HSZ + e]);
        }
        if (wid == 0) __builtin_amdgcn_s_wait_tensorcnt(0);
        __syncthreads();
#pragma unroll
        for (int kk = 0; kk < 64; kk += 32) {
            Frag a;
            const int ar = wid * 16 + lo;
            a.q[0] = *(const uint4*)&lsA[ar * 80 + kk + hi * 8];
            a.q[1] = *(const uint4*)&lsA[ar * 80 + kk + hi * 8 + 16];
            Frag b[4];
#pragma unroll
            for (int n = 0; n < 4; ++n) {
                const int br = n * 16 + lo;
                b[n].q[0] = *(const uint4*)&lsB[br * 80 + kk + hi * 8];
                b[n].q[1] = *(const uint4*)&lsB[br * 80 + kk + hi * 8 + 16];
            }
#pragma unroll
            for (int n = 0; n < 4; ++n) acc[n] = wmma_bf16(a, b[n], acc[n]);
        }
        __syncthreads();
    }
    // store bf16 into [B,H,S,HS]
#pragma unroll
    for (int n = 0; n < 4; ++n) {
        const int e = n * 16 + lo;
#pragma unroll
        for (int r = 0; r < 8; ++r) {
            const int grow = rowBase + wid * 16 + r + 8 * hi; // flat (b,s)
            const int b = grow >> 11;          // / 2048
            const int s = grow & 2047;
            Dst[(((size_t)b * HH + h) * SS + s) * HSZ + e] = f2bf(acc[n][r]);
        }
    }
}

// ---------------------------------------------------------------------------
// Kernel 3: causal flash attention per (b, h, 64-query tile), 4 waves
// K tile staged by TDM; V tile transposed manually (TDM has no transpose).
// ---------------------------------------------------------------------------
__global__ void __launch_bounds__(128)
attn_kernel(const unsigned short* __restrict__ qh, const unsigned short* __restrict__ kh,
            const unsigned short* __restrict__ vh, unsigned short* __restrict__ ao) {
    const int qt = blockIdx.x;                 // query tile (64 rows)
    const int h  = blockIdx.y;
    const int b  = blockIdx.z;
    const int qb = qt * 64;
    const int tid = threadIdx.x;
    const int wid = tid >> 5, lane = tid & 31;
    const int lo = lane & 15, hi = lane >> 4;

    const size_t headBase = ((size_t)b * HH + h) * SS;
    const unsigned short* Q = qh + headBase * HSZ;
    const unsigned short* K = kh + headBase * HSZ;
    const unsigned short* V = vh + headBase * HSZ;

    __shared__ unsigned short lsK[64 * 80];       // K tile [key][dim] (TDM-padded)
    __shared__ unsigned short lsVt[64 * 80];      // V tile transposed [dim][key]
    __shared__ unsigned short lsP[4][16 * 80];    // per-wave P staging [qrow][key]

    // Q fragments held in registers for the whole kernel (K-dim = 64 -> 2 frags)
    Frag qa[2];
    {
        const unsigned short* qp = Q + (size_t)(qb + wid * 16 + lo) * HSZ;
#pragma unroll
        for (int kk = 0; kk < 2; ++kk) {
            qa[kk].q[0] = *(const uint4*)&qp[kk * 32 + hi * 8];
            qa[kk].q[1] = *(const uint4*)&qp[kk * 32 + hi * 8 + 16];
        }
    }

    v8f acc[4];
    float m8[8], l8[8];
#pragma unroll
    for (int n = 0; n < 4; ++n)
#pragma unroll
        for (int i = 0; i < 8; ++i) acc[n][i] = 0.f;
#pragma unroll
    for (int r = 0; r < 8; ++r) { m8[r] = -1e30f; l8[r] = 0.f; }

    for (int j = 0; j <= qt; ++j) {
        const int kb = j * 64;
        // async DMA of the 64x64 K tile
        if (wid == 0)
            tdm_load_tile_bf16((unsigned)(uintptr_t)lsK,
                               K + (size_t)kb * HSZ, 64, 64, HSZ, true);
        if (j < qt)
            __builtin_prefetch(&V[(size_t)(kb + 64 + (tid >> 1)) * HSZ], 0, 3);
        // stage V transposed: vh[kb+key][e] -> lsVt[e][key]
        for (int idx = tid; idx < 4096; idx += 128) {
            const int e = idx & 63, key = idx >> 6;
            lsVt[e * 80 + key] = V[(size_t)(kb + key) * HSZ + e];
        }
        if (wid == 0) __builtin_amdgcn_s_wait_tensorcnt(0);
        __syncthreads();

        // S = Q K^T  (16 q-rows x 64 keys per wave)
        v8f sf[4];
#pragma unroll
        for (int n = 0; n < 4; ++n) {
            v8f z;
#pragma unroll
            for (int i = 0; i < 8; ++i) z[i] = 0.f;
            Frag bk[2];
#pragma unroll
            for (int kk = 0; kk < 2; ++kk) {
                const int br = n * 16 + lo;
                bk[kk].q[0] = *(const uint4*)&lsK[br * 80 + kk * 32 + hi * 8];
                bk[kk].q[1] = *(const uint4*)&lsK[br * 80 + kk * 32 + hi * 8 + 16];
            }
            z = wmma_bf16(qa[0], bk[0], z);
            z = wmma_bf16(qa[1], bk[1], z);
            sf[n] = z;
        }

        // online softmax (row = r + 8*hi lives on one 16-lane half)
        const bool diag = (j == qt);
#pragma unroll
        for (int r = 0; r < 8; ++r) {
            const int qrow = qb + wid * 16 + r + 8 * hi;
            float mloc = -1e30f;
#pragma unroll
            for (int n = 0; n < 4; ++n) {
                float sv = sf[n][r] * ATT_SCALE;
                if (diag && (kb + n * 16 + lo) > qrow) sv = -1e30f;
                sf[n][r] = sv;
                mloc = fmaxf(mloc, sv);
            }
            for (int i = 1; i < 16; i <<= 1) mloc = fmaxf(mloc, __shfl_xor(mloc, i, 32));
            const float mnew  = fmaxf(m8[r], mloc);
            const float alpha = __expf(m8[r] - mnew);
            float ps = 0.f;
#pragma unroll
            for (int n = 0; n < 4; ++n) {
                const float p = __expf(sf[n][r] - mnew);
                sf[n][r] = p;
                ps += p;
            }
            for (int i = 1; i < 16; i <<= 1) ps += __shfl_xor(ps, i, 32);
            l8[r] = l8[r] * alpha + ps;
            m8[r] = mnew;
#pragma unroll
            for (int n = 0; n < 4; ++n) acc[n][r] *= alpha;
        }

        // C-layout -> A-layout for P via per-wave LDS region
#pragma unroll
        for (int n = 0; n < 4; ++n)
#pragma unroll
            for (int r = 0; r < 8; ++r)
                lsP[wid][(r + 8 * hi) * 80 + n * 16 + lo] = f2bf(sf[n][r]);
        __syncthreads();

        // O += P V
#pragma unroll
        for (int kk = 0; kk < 2; ++kk) {
            Frag pa;
            pa.q[0] = *(const uint4*)&lsP[wid][lo * 80 + kk * 32 + hi * 8];
            pa.q[1] = *(const uint4*)&lsP[wid][lo * 80 + kk * 32 + hi * 8 + 16];
            Frag bv[4];
#pragma unroll
            for (int n = 0; n < 4; ++n) {
                const int er = n * 16 + lo;
                bv[n].q[0] = *(const uint4*)&lsVt[er * 80 + kk * 32 + hi * 8];
                bv[n].q[1] = *(const uint4*)&lsVt[er * 80 + kk * 32 + hi * 8 + 16];
            }
#pragma unroll
            for (int n = 0; n < 4; ++n) acc[n] = wmma_bf16(pa, bv[n], acc[n]);
        }
        __syncthreads();
    }

    // normalize and store concat-head layout ao[b][s][h*64 + e] (bf16)
#pragma unroll
    for (int r = 0; r < 8; ++r) {
        const float inv = 1.0f / l8[r];
        const int s = qb + wid * 16 + r + 8 * hi;
#pragma unroll
        for (int n = 0; n < 4; ++n) {
            const int e = h * HSZ + n * 16 + lo;
            ao[((size_t)b * SS + s) * DD + e] = f2bf(acc[n][r] * inv);
        }
    }
}

// ---------------------------------------------------------------------------
// Kernel 4: output projection  out = ao @ Wp^T + bp   (fp32 out)
// A-tile staged by TDM; Wp rows are already the B^T layout (VALU convert).
// ---------------------------------------------------------------------------
__global__ void __launch_bounds__(256)
outproj_kernel(const unsigned short* __restrict__ ao, const float* __restrict__ Wp,
               const float* __restrict__ bp, float* __restrict__ out) {
    const int rowBase = blockIdx.x * 128;
    const int nBase   = blockIdx.y * 128;
    const int tid = threadIdx.x;
    const int wid = tid >> 5, lane = tid & 31;
    const int lo = lane & 15, hi = lane >> 4;

    __shared__ unsigned short lsA[128 * 80];   // activations [row][k] (TDM-padded)
    __shared__ unsigned short lsB[128 * 80];   // Wp rows [n][k]

    v8f acc[8];
#pragma unroll
    for (int n = 0; n < 8; ++n)
#pragma unroll
        for (int i = 0; i < 8; ++i) acc[n][i] = 0.f;

    for (int kb = 0; kb < DD; kb += 64) {
        if (wid == 0)
            tdm_load_tile_bf16((unsigned)(uintptr_t)lsA,
                               ao + (size_t)rowBase * DD + kb,
                               64, 128, DD, true);
        for (int idx = tid; idx < 8192; idx += 256) {
            const int d = idx & 63, n = idx >> 6;
            lsB[n * 80 + d] = f2bf(Wp[(size_t)(nBase + n) * DD + kb + d]);
        }
        if (wid == 0) __builtin_amdgcn_s_wait_tensorcnt(0);
        __syncthreads();
#pragma unroll
        for (int kk = 0; kk < 64; kk += 32) {
            Frag a;
            const int ar = wid * 16 + lo;
            a.q[0] = *(const uint4*)&lsA[ar * 80 + kk + hi * 8];
            a.q[1] = *(const uint4*)&lsA[ar * 80 + kk + hi * 8 + 16];
#pragma unroll
            for (int n = 0; n < 8; ++n) {
                Frag bfr;
                const int br = n * 16 + lo;
                bfr.q[0] = *(const uint4*)&lsB[br * 80 + kk + hi * 8];
                bfr.q[1] = *(const uint4*)&lsB[br * 80 + kk + hi * 8 + 16];
                acc[n] = wmma_bf16(a, bfr, acc[n]);
            }
        }
        __syncthreads();
    }
#pragma unroll
    for (int n = 0; n < 8; ++n) {
        const int col = nBase + n * 16 + lo;
        const float bias = bp[col];
#pragma unroll
        for (int r = 0; r < 8; ++r) {
            const int row = rowBase + wid * 16 + r + 8 * hi;
            out[(size_t)row * DD + col] = acc[n][r] + bias;
        }
    }
}

// ---------------------------------------------------------------------------
extern "C" void kernel_launch(void* const* d_in, const int* in_sizes, int n_in,
                              void* d_out, int out_size, void* d_ws, size_t ws_size,
                              hipStream_t stream) {
    (void)in_sizes; (void)n_in; (void)out_size; (void)ws_size;
    const float* v    = (const float*)d_in[0];
    const float* k    = (const float*)d_in[1];
    const float* q    = (const float*)d_in[2];
    /* d_in[3] = causal mask (structure known statically; unused) */
    const float* ln_g = (const float*)d_in[4];
    const float* ln_b = (const float*)d_in[5];
    const float* Wq   = (const float*)d_in[6];
    const float* Wk   = (const float*)d_in[7];
    const float* Wv   = (const float*)d_in[8];
    const float* Wp   = (const float*)d_in[9];
    const float* bp   = (const float*)d_in[10];
    float* out = (float*)d_out;

    char* ws = (char*)d_ws;
    const size_t SZ = (size_t)BB * SS * DD * sizeof(unsigned short);  // 16 MB each
    unsigned short* qn = (unsigned short*)(ws + 0 * SZ);
    unsigned short* kn = (unsigned short*)(ws + 1 * SZ);
    unsigned short* vn = (unsigned short*)(ws + 2 * SZ);
    unsigned short* qh = (unsigned short*)(ws + 3 * SZ);
    unsigned short* kh = (unsigned short*)(ws + 4 * SZ);
    unsigned short* vh = (unsigned short*)(ws + 5 * SZ);
    unsigned short* ao = (unsigned short*)(ws + 6 * SZ);

    ln_kernel<<<dim3(BB * SS, 3), 256, 0, stream>>>(v, k, q, ln_g, ln_b, vn, kn, qn);
    proj_kernel<<<dim3(BB * SS / 128, HH, 3), 256, 0, stream>>>(qn, kn, vn,
                                                                Wq, Wk, Wv,
                                                                qh, kh, vh);
    attn_kernel<<<dim3(SS / 64, HH, BB), 128, 0, stream>>>(qh, kh, vh, ao);
    outproj_kernel<<<dim3(BB * SS / 128, DD / 128), 256, 0, stream>>>(ao, Wp, bp, out);
}
